// LinearNO_block_2765958939230
// MI455X (gfx1250) — matrix operand order repaired and checked by reading, not compiled, not run on hardware
//
#include <hip/hip_runtime.h>
#include <hip/hip_bf16.h>

// ---------------------------------------------------------------------------
// Problem constants (from the reference)
// ---------------------------------------------------------------------------
constexpr int    BB    = 8;
constexpr int    NN    = 8192;
constexpr int    CC    = 256;
constexpr int    HEADS = 8;
constexpr int    DH    = 32;
constexpr int    KR    = 4;
constexpr int    INNER = 256;   // HEADS*DH
constexpr int    FF    = 1024;  // 4*C
constexpr size_t TOK   = (size_t)BB * NN;       // 65536 tokens
constexpr size_t ROWS  = TOK * HEADS;           // 524288 (token,head) rows

typedef __attribute__((ext_vector_type(16))) __bf16 v16bf;
typedef __attribute__((ext_vector_type(8)))  float  v8f;

static __device__ __forceinline__ unsigned short f2bf(float f) {
  unsigned int u = __builtin_bit_cast(unsigned int, f);
  unsigned int r = u + 0x7FFFu + ((u >> 16) & 1u);   // round-to-nearest-even
  return (unsigned short)(r >> 16);
}
static __device__ __forceinline__ float bf2f(unsigned short h) {
  unsigned int u = ((unsigned int)h) << 16;
  return __builtin_bit_cast(float, u);
}

// ---------------------------------------------------------------------------
// Weight prep: f32 [K,N] row-major  ->  bf16 transposed [N,K]
// ---------------------------------------------------------------------------
__global__ void prep_wt_kernel(const float* __restrict__ W,
                               unsigned short* __restrict__ Wt,
                               int K, int N) {
  int t = blockIdx.x * 256 + threadIdx.x;
  if (t >= K * N) return;
  int n = t / K, k = t % K;
  Wt[(size_t)n * K + k] = f2bf(W[(size_t)k * N + n]);
}

// Concatenated small-weights matrix Bcat[K=32, N=48] = [Wv | Wq | Wk | pad0],
// stored transposed bf16 [48,32].
__global__ void prep_bcat_kernel(const float* __restrict__ Wq,
                                 const float* __restrict__ Wk,
                                 const float* __restrict__ Wv,
                                 unsigned short* __restrict__ Bt) {
  int t = blockIdx.x * 256 + threadIdx.x;
  if (t >= 48 * 32) return;
  int n = t >> 5, k = t & 31;
  float v = 0.0f;
  if (n < 32)      v = Wv[k * DH + n];
  else if (n < 36) v = Wq[k * KR + (n - 32)];
  else if (n < 40) v = Wk[k * KR + (n - 36)];
  Bt[n * 32 + k] = f2bf(v);
}

// ---------------------------------------------------------------------------
// LayerNorm over C=256, f32 in -> bf16 out.  One wave per token (8 waves/block)
// ---------------------------------------------------------------------------
__global__ __launch_bounds__(256)
void ln_kernel(const float* __restrict__ x, const float* __restrict__ g,
               const float* __restrict__ beta, unsigned short* __restrict__ out) {
  const int wave = threadIdx.x >> 5, lane = threadIdx.x & 31;
  const size_t token = (size_t)blockIdx.x * 8 + wave;
  const float4* r4 = (const float4*)(x + token * CC) + lane * 2;
  float4 a = r4[0], c = r4[1];
  float v[8] = {a.x, a.y, a.z, a.w, c.x, c.y, c.z, c.w};

  float s = 0.f;
#pragma unroll
  for (int j = 0; j < 8; ++j) s += v[j];
#pragma unroll
  for (int m = 16; m > 0; m >>= 1) s += __shfl_xor(s, m);
  const float mu = s * (1.0f / 256.0f);

  float q = 0.f;
#pragma unroll
  for (int j = 0; j < 8; ++j) { v[j] -= mu; q += v[j] * v[j]; }
#pragma unroll
  for (int m = 16; m > 0; m >>= 1) q += __shfl_xor(q, m);
  const float rstd = rsqrtf(q * (1.0f / 256.0f) + 1e-5f);

  const float4* g4 = (const float4*)g + lane * 2;
  const float4* b4 = (const float4*)beta + lane * 2;
  float4 ga = g4[0], gb = g4[1], ba = b4[0], bb = b4[1];
  float gg[8] = {ga.x, ga.y, ga.z, ga.w, gb.x, gb.y, gb.z, gb.w};
  float bv[8] = {ba.x, ba.y, ba.z, ba.w, bb.x, bb.y, bb.z, bb.w};

  unsigned int p[4];
#pragma unroll
  for (int j = 0; j < 4; ++j) {
    float o0 = v[2 * j]     * rstd * gg[2 * j]     + bv[2 * j];
    float o1 = v[2 * j + 1] * rstd * gg[2 * j + 1] + bv[2 * j + 1];
    p[j] = (unsigned int)f2bf(o0) | ((unsigned int)f2bf(o1) << 16);
  }
  uint4 pk; pk.x = p[0]; pk.y = p[1]; pk.z = p[2]; pk.w = p[3];
  *(uint4*)(out + token * CC + lane * 8) = pk;
}

// ---------------------------------------------------------------------------
// Generic bf16 WMMA GEMM (K fixed = 256, N-slab of 64 per block):
//   C[M, ...] = A[M,256](bf16) * B[256, N](bf16, stored Bt[N, ldb])
// Block: 256 threads = 8 waves; wave w handles rows [blk.x*256 + w*32, +32)
// (two 16-row A fragments, 8 accumulators) and the full 64-column slab.
// B panel (64 x 256 bf16) is staged to LDS with GLOBAL_LOAD_ASYNC_TO_LDS_B128
// (ASYNCcnt tracked, waited with s_wait_asynccnt) -- CDNA5 async copy path.
// Grid = (M/256, N/64).
// ---------------------------------------------------------------------------
enum { EPI_F32 = 1, EPI_BF16 = 2, EPI_BIAS = 4, EPI_RESID = 8,
       EPI_GELU = 16, EPI_ACCUM = 32 };

template <int EPI>
__global__ __launch_bounds__(256)
void wmma_gemm_bf16(const unsigned short* __restrict__ A, int lda,
                    const unsigned short* __restrict__ Bt, int ldb,
                    const float* __restrict__ bias,
                    const float* __restrict__ resid,
                    float* __restrict__ Cf,
                    unsigned short* __restrict__ Cb,
                    int ldc) {
  constexpr int K       = 256;
  constexpr int BSTRIDE = K + 8;                 // 264 ushorts = 528 B (16B-aligned)
  __shared__ unsigned short bs[64 * BSTRIDE];    // ~33 KB

  const int tid  = threadIdx.x;
  const int wave = tid >> 5;
  const int lane = tid & 31;
  const int half = lane >> 4;                    // K-group select per ISA layout
  const int l16  = lane & 15;
  const int row0  = blockIdx.x * 256 + wave * 32;
  const int ncol0 = blockIdx.y * 64;

  // ---- async stage of the 64x256 B panel into LDS (memory -> LDS direct) ----
  // 64 rows * 32 uint4 per row = 2048 b128 transfers; 8 per thread, coalesced.
#pragma unroll
  for (int it = 0; it < 8; ++it) {
    const int i  = tid + it * 256;
    const int r  = i >> 5;                       // row of Bt panel (0..63)
    const int kk = (i & 31) * 8;                 // ushort offset within row
    const unsigned lds_off = (unsigned)(uintptr_t)(bs + r * BSTRIDE + kk);
    const unsigned short* gp = Bt + (size_t)(ncol0 + r) * ldb + kk;
    asm volatile("global_load_async_to_lds_b128 %0, %1, off"
                 :: "v"(lds_off), "v"(gp) : "memory");
  }
  asm volatile("s_wait_asynccnt 0x0" ::: "memory");
  __syncthreads();

  const unsigned short* arow0 = A + (size_t)(row0 + l16) * lda + half * 8;
  const unsigned short* arow1 = arow0 + (size_t)16 * lda;

  v8f acc[2][4] = {{{}, {}, {}, {}}, {{}, {}, {}, {}}};
  for (int k = 0; k < K; k += 32) {
    __builtin_prefetch(arow0 + k + 64, 0, 0);    // global_prefetch_b8
    union { uint4 u[2]; v16bf v; } fa0, fa1;
    fa0.u[0] = *(const uint4*)(arow0 + k);       // A: K = k + half*8 + 0..7
    fa0.u[1] = *(const uint4*)(arow0 + k + 16);  //    K = k + 16 + half*8 + 0..7
    fa1.u[0] = *(const uint4*)(arow1 + k);
    fa1.u[1] = *(const uint4*)(arow1 + k + 16);
#pragma unroll
    for (int nt = 0; nt < 4; ++nt) {
      union { uint4 u[2]; v16bf v; } fb;
      const unsigned short* bp = bs + (nt * 16 + l16) * BSTRIDE + half * 8 + k;
      fb.u[0] = *(const uint4*)(bp);             // B: K = k + half*8 + 0..7
      fb.u[1] = *(const uint4*)(bp + 16);
      acc[0][nt] = __builtin_amdgcn_wmma_f32_16x16x32_bf16(
          false, fa0.v, false, fb.v, (short)0, acc[0][nt], false, false);
      acc[1][nt] = __builtin_amdgcn_wmma_f32_16x16x32_bf16(
          false, fa1.v, false, fb.v, (short)0, acc[1][nt], false, false);
    }
  }

#pragma unroll
  for (int mt = 0; mt < 2; ++mt) {
#pragma unroll
    for (int nt = 0; nt < 4; ++nt) {
      const int col = ncol0 + nt * 16 + l16;
      const float bv = (EPI & EPI_BIAS) ? bias[col] : 0.0f;
#pragma unroll
      for (int i = 0; i < 8; ++i) {
        const int row = row0 + mt * 16 + half * 8 + i;   // C: VGPR i -> M=i / i+8
        const size_t idx = (size_t)row * ldc + col;
        float v = acc[mt][nt][i] + bv;
        if (EPI & EPI_GELU)  v = 0.5f * v * (1.0f + erff(v * 0.70710678118f));
        if (EPI & EPI_RESID) v += resid[idx];
        if (EPI & EPI_ACCUM) v += Cf[idx];
        if (EPI & EPI_F32)   Cf[idx] = v;
        if (EPI & EPI_BF16)  Cb[idx] = f2bf(v);
      }
    }
  }
}

// ---------------------------------------------------------------------------
// Fused small GEMM: xm[ROWS,32](bf16) @ Bcat[32,48] -> v (bf16) + q/k logits(f32)
// Single 16x16x32 WMMA per tile. grid = (ROWS/128, 3).
// ---------------------------------------------------------------------------
__global__ __launch_bounds__(256)
void wmma_vqk_kernel(const unsigned short* __restrict__ A,
                     const unsigned short* __restrict__ Bt,
                     unsigned short* __restrict__ Vout,    // [ROWS,32] bf16
                     float* __restrict__ SQK) {            // [ROWS,16] f32
  const int wave = threadIdx.x >> 5;
  const int lane = threadIdx.x & 31;
  const int half = lane >> 4, l16 = lane & 15;
  const int row0 = (blockIdx.x * 8 + wave) * 16;
  const int col0 = blockIdx.y * 16;

  const unsigned short* arow = A  + (size_t)(row0 + l16) * 32 + half * 8;
  const unsigned short* brow = Bt + (size_t)(col0 + l16) * 32 + half * 8;

  union { uint4 u[2]; v16bf v; } fa, fb;
  fa.u[0] = *(const uint4*)(arow);
  fa.u[1] = *(const uint4*)(arow + 16);
  fb.u[0] = *(const uint4*)(brow);
  fb.u[1] = *(const uint4*)(brow + 16);
  v8f acc = {};
  acc = __builtin_amdgcn_wmma_f32_16x16x32_bf16(false, fa.v, false, fb.v,
                                                (short)0, acc, false, false);
  const int col = col0 + l16;
  if (blockIdx.y < 2) {
#pragma unroll
    for (int i = 0; i < 8; ++i) {
      const int row = row0 + half * 8 + i;
      Vout[(size_t)row * 32 + col] = f2bf(acc[i]);
    }
  } else {
#pragma unroll
    for (int i = 0; i < 8; ++i) {
      const int row = row0 + half * 8 + i;
      SQK[(size_t)row * 16 + (col - 32)] = acc[i];
    }
  }
}

// ---------------------------------------------------------------------------
// Per-(b,h,r) max over n of the k-logits.  256 blocks.
// ---------------------------------------------------------------------------
__global__ __launch_bounds__(256)
void kmax_kernel(const float* __restrict__ SQK, float* __restrict__ kmax) {
  const int g = blockIdx.x;                 // ((b*8+h)*4+r)
  const int b = g >> 5, h = (g >> 2) & 7, r = g & 3;
  float m = -1e30f;
  for (int n = threadIdx.x; n < NN; n += 256) {
    const size_t row = ((size_t)(b * NN + n)) * HEADS + h;
    m = fmaxf(m, SQK[row * 16 + 4 + r]);
  }
  __shared__ float red[256];
  red[threadIdx.x] = m;
  __syncthreads();
  for (int s = 128; s > 0; s >>= 1) {
    if (threadIdx.x < s) red[threadIdx.x] = fmaxf(red[threadIdx.x], red[threadIdx.x + s]);
    __syncthreads();
  }
  if (threadIdx.x == 0) kmax[g] = red[0];
}

// ---------------------------------------------------------------------------
// kv[b,h,r,d] = sum_n softmax_n(k)[n,r] * v[n,d].   64 blocks (one per b,h).
// ---------------------------------------------------------------------------
__global__ __launch_bounds__(256)
void kv_kernel(const float* __restrict__ SQK, const float* __restrict__ kmax,
               const unsigned short* __restrict__ vb, float* __restrict__ kvout) {
  __shared__ float lds_kv[8][4][32];
  __shared__ float lds_d[8][4];
  __shared__ float denom_sh[4];
  const int g = blockIdx.x;                  // b*8+h
  const int b = g >> 3, h = g & 7;
  const int t = threadIdx.x, d = t & 31, s = t >> 5;

  float mx[4];
#pragma unroll
  for (int r = 0; r < 4; ++r) mx[r] = kmax[(g << 2) + r];

  float acc[4] = {0.f, 0.f, 0.f, 0.f};
  float den[4] = {0.f, 0.f, 0.f, 0.f};
  for (int n = s; n < NN; n += 8) {
    const size_t row = ((size_t)(b * NN + n)) * HEADS + h;
    const float* sk = SQK + row * 16 + 4;
    const float vv = bf2f(vb[row * 32 + d]);
#pragma unroll
    for (int r = 0; r < 4; ++r) {
      const float e = __expf(sk[r] - mx[r]);
      acc[r] += e * vv;
      if (d == 0) den[r] += e;
    }
  }
#pragma unroll
  for (int r = 0; r < 4; ++r) lds_kv[s][r][d] = acc[r];
  if (d == 0) {
#pragma unroll
    for (int r = 0; r < 4; ++r) lds_d[s][r] = den[r];
  }
  __syncthreads();
  if (t < 4) {
    float dd = 0.f;
    for (int ss = 0; ss < 8; ++ss) dd += lds_d[ss][t];
    denom_sh[t] = dd;
  }
  __syncthreads();
  if (t < 128) {
    const int r = t >> 5, dd = t & 31;
    float sum = 0.f;
    for (int ss = 0; ss < 8; ++ss) sum += lds_kv[ss][r][dd];
    kvout[((size_t)g * 4 + r) * 32 + dd] = sum / denom_sh[r];
  }
}

// ---------------------------------------------------------------------------
// qkv[token,h,d] = softmax_r(q)[r] . kv[b,h,r,d]  -> bf16.  One block per token.
// ---------------------------------------------------------------------------
__global__ __launch_bounds__(256)
void qkv_kernel(const float* __restrict__ SQK, const float* __restrict__ kv,
                unsigned short* __restrict__ out) {
  const size_t token = blockIdx.x;
  const int t = threadIdx.x, h = t >> 5, d = t & 31;
  const size_t row = token * HEADS + h;
  const float* sq = SQK + row * 16;
  const float a0 = sq[0], a1 = sq[1], a2 = sq[2], a3 = sq[3];
  const float mx = fmaxf(fmaxf(a0, a1), fmaxf(a2, a3));
  const float e0 = __expf(a0 - mx), e1 = __expf(a1 - mx);
  const float e2 = __expf(a2 - mx), e3 = __expf(a3 - mx);
  const float inv = 1.0f / (e0 + e1 + e2 + e3);
  const int b = (int)(token >> 13);
  const float* kvp = kv + ((size_t)(b * 8 + h) * 4) * 32 + d;
  const float val = (e0 * kvp[0] + e1 * kvp[32] + e2 * kvp[64] + e3 * kvp[96]) * inv;
  out[token * INNER + h * DH + d] = f2bf(val);
}

// ---------------------------------------------------------------------------
// Host-side orchestration
// ---------------------------------------------------------------------------
extern "C" void kernel_launch(void* const* d_in, const int* in_sizes, int n_in,
                              void* d_out, int out_size, void* d_ws, size_t ws_size,
                              hipStream_t stream) {
  (void)in_sizes; (void)n_in; (void)out_size; (void)ws_size;
  const float* x     = (const float*)d_in[0];
  const float* ln1_g = (const float*)d_in[1];
  const float* ln1_b = (const float*)d_in[2];
  const float* Wx    = (const float*)d_in[3];
  const float* bx    = (const float*)d_in[4];
  const float* Wq    = (const float*)d_in[5];
  const float* Wk    = (const float*)d_in[6];
  const float* Wv    = (const float*)d_in[7];
  const float* Wo    = (const float*)d_in[8];
  const float* bo    = (const float*)d_in[9];
  const float* ln2_g = (const float*)d_in[10];
  const float* ln2_b = (const float*)d_in[11];
  const float* W1    = (const float*)d_in[12];
  const float* b1    = (const float*)d_in[13];
  const float* W2    = (const float*)d_in[14];
  const float* b2    = (const float*)d_in[15];
  float* outp = (float*)d_out;

  // ---- workspace carve (aligned to 256B) ----
  char* ws = (char*)d_ws;
  size_t off = 0;
  auto carve = [&](size_t bytes) -> char* {
    char* p = ws + off;
    off = (off + bytes + 255) & ~(size_t)255;
    return p;
  };
  unsigned short* buf_h  = (unsigned short*)carve(TOK * CC * 2);     // LN out / qkv / h2
  unsigned short* buf_xm = (unsigned short*)carve(TOK * INNER * 2);  // xm / mlp hidden chunk
  unsigned short* buf_v  = (unsigned short*)carve(ROWS * DH * 2);    // v
  float*          buf_sqk= (float*)carve(ROWS * 16 * 4);             // q/k logits
  float*          buf_fx = (float*)carve(TOK * CC * 4);              // fx (f32)
  unsigned short* wx_t   = (unsigned short*)carve((size_t)CC * INNER * 2);
  unsigned short* wo_t   = (unsigned short*)carve((size_t)INNER * CC * 2);
  unsigned short* w1_t   = (unsigned short*)carve((size_t)FF * CC * 2);
  unsigned short* w2_t   = (unsigned short*)carve((size_t)CC * FF * 2);
  unsigned short* bcat   = (unsigned short*)carve(48 * 32 * 2);
  float*          kmaxb  = (float*)carve(256 * 4);
  float*          kvb    = (float*)carve(64 * 4 * 32 * 4);

  // ---- weight prep ----
  prep_wt_kernel<<<(CC * INNER + 255) / 256, 256, 0, stream>>>(Wx, wx_t, CC, INNER);
  prep_wt_kernel<<<(INNER * CC + 255) / 256, 256, 0, stream>>>(Wo, wo_t, INNER, CC);
  prep_wt_kernel<<<(CC * FF + 255) / 256, 256, 0, stream>>>(W1, w1_t, CC, FF);
  prep_wt_kernel<<<(FF * CC + 255) / 256, 256, 0, stream>>>(W2, w2_t, FF, CC);
  prep_bcat_kernel<<<6, 256, 0, stream>>>(Wq, Wk, Wv, bcat);

  const dim3 blk(256);
  const dim3 gemm_grid(TOK / 256, 4);             // 256-row blocks, 64-col slabs

  // ---- LN1 -> h (bf16) ----
  ln_kernel<<<TOK / 8, blk, 0, stream>>>(x, ln1_g, ln1_b, buf_h);

  // ---- xm = h @ Wx + bx  (bf16 out) ----
  wmma_gemm_bf16<EPI_BF16 | EPI_BIAS><<<gemm_grid, blk, 0, stream>>>(
      buf_h, CC, wx_t, CC, bx, nullptr, nullptr, buf_xm, INNER);

  // ---- v / q-logits / k-logits in one padded WMMA GEMM ----
  wmma_vqk_kernel<<<dim3(ROWS / 128, 3), blk, 0, stream>>>(buf_xm, bcat, buf_v, buf_sqk);

  // ---- k softmax over n + kv accumulation ----
  kmax_kernel<<<256, blk, 0, stream>>>(buf_sqk, kmaxb);
  kv_kernel<<<64, blk, 0, stream>>>(buf_sqk, kmaxb, buf_v, kvb);

  // ---- q softmax (over r) and qkv = q @ kv  (bf16, overwrite buf_h) ----
  qkv_kernel<<<(unsigned)TOK, blk, 0, stream>>>(buf_sqk, kvb, buf_h);

  // ---- fx = qkv @ Wo + bo + x  (f32) ----
  wmma_gemm_bf16<EPI_F32 | EPI_BIAS | EPI_RESID><<<gemm_grid, blk, 0, stream>>>(
      buf_h, INNER, wo_t, INNER, bo, x, buf_fx, nullptr, CC);

  // ---- LN2 -> h2 (bf16, overwrite buf_h) ----
  ln_kernel<<<TOK / 8, blk, 0, stream>>>(buf_fx, ln2_g, ln2_b, buf_h);

  // ---- MLP in 4 FF-chunks of 256 (hidden chunk reuses buf_xm) ----
  for (int j = 0; j < 4; ++j) {
    // hidden_j = gelu(h2 @ W1[:, j*256 : j*256+256] + b1_j)   (bf16)
    wmma_gemm_bf16<EPI_BF16 | EPI_BIAS | EPI_GELU><<<gemm_grid, blk, 0, stream>>>(
        buf_h, CC, w1_t + (size_t)j * 256 * CC, CC, b1 + j * 256,
        nullptr, nullptr, buf_xm, 256);
    if (j == 0) {
      // out = hidden_0 @ W2[0:256,:] + b2 + fx
      wmma_gemm_bf16<EPI_F32 | EPI_BIAS | EPI_RESID><<<gemm_grid, blk, 0, stream>>>(
          buf_xm, 256, w2_t + (size_t)j * 256, FF, b2, buf_fx, outp, nullptr, CC);
    } else {
      // out += hidden_j @ W2[j*256 : j*256+256, :]
      wmma_gemm_bf16<EPI_F32 | EPI_ACCUM><<<gemm_grid, blk, 0, stream>>>(
          buf_xm, 256, w2_t + (size_t)j * 256, FF, nullptr, nullptr, outp, nullptr, CC);
    }
  }
}
